// GraphAttention_33861522162300
// MI455X (gfx1250) — matrix operand order repaired and checked by reading, not compiled, not run on hardware
//
#include <hip/hip_runtime.h>
#include <hip/hip_bf16.h>

typedef __attribute__((ext_vector_type(16))) _Float16 v16h;
typedef __attribute__((ext_vector_type(8)))  float    v8f;

#define N_NODES 8192
#define F_IN    256
#define F_OUT   128
#define ALPHA   0.2f
#define JSPLIT  4
#define JCHUNK  (N_NODES / JSPLIT)

// ---------------------------------------------------------------------------
// Pack W [256][128] f32 -> WMMA B-layout f16 tiles.
// WB[ks][t][lane][v] : uint32 = pack(f16(W[32ks+kb+2v+(v>=4?8:0)][16t+n]), f16(W[...+1][...]))
// kb = (lane>=16)?8:0, n = lane&15.   (B 32x16 f16: lanes0-15 K=0..7,16..23, lanes16-31 K=8..15,24..31)
// ---------------------------------------------------------------------------
__global__ __launch_bounds__(256) void packW_kernel(const float* __restrict__ W,
                                                    unsigned int* __restrict__ WB) {
    int tid  = blockIdx.x * 256 + threadIdx.x;          // 16384 total
    int v    = tid & 7;
    int lane = (tid >> 3) & 31;
    int t    = (tid >> 8) & 7;
    int ks   = tid >> 11;
    int n    = lane & 15;
    int kb   = (lane >= 16) ? 8 : 0;
    int row  = 32 * ks + kb + 2 * v + (v >= 4 ? 8 : 0);
    int col  = 16 * t + n;
    _Float16 h0 = (_Float16)W[row * F_OUT + col];
    _Float16 h1 = (_Float16)W[(row + 1) * F_OUT + col];
    unsigned int u0 = (unsigned int)__builtin_bit_cast(unsigned short, h0);
    unsigned int u1 = (unsigned int)__builtin_bit_cast(unsigned short, h1);
    WB[tid] = u0 | (u1 << 16);
}

// ---------------------------------------------------------------------------
// h = input @ W  (8192x256x128), f32 accum via v_wmma_f32_16x16x32_f16.
// One wave computes 16 rows x 128 cols; 64 blocks x 8 waves cover all rows.
// sched_barrier(0) pins the load group above the compute group so B tiles
// stay in distinct registers and waits are incremental.
// ---------------------------------------------------------------------------
__global__ __launch_bounds__(256) __attribute__((amdgpu_waves_per_eu(2)))
void h_kernel(const float* __restrict__ input,
              const v16h* __restrict__ WB,
              float* __restrict__ h) {
    const int lane    = threadIdx.x & 31;
    const int wave    = threadIdx.x >> 5;
    const int rowBase = blockIdx.x * 128 + wave * 16;
    const int m       = lane & 15;
    const bool hi     = lane >= 16;
    const int kb0     = hi ? 8 : 0;

    const float* inRow = input + (size_t)(rowBase + m) * F_IN;

    v8f c[8] = {};
    #pragma unroll
    for (int ks = 0; ks < 8; ++ks) {
        // ---- load group ----
        v16h bt[8];
        #pragma unroll
        for (int t = 0; t < 8; ++t) bt[t] = WB[(ks * 8 + t) * 32 + lane];

        const float4* p0 = (const float4*)(inRow + ks * 32 + kb0);
        const float4* p1 = (const float4*)(inRow + ks * 32 + kb0 + 16);
        float4 X0 = p0[0], X1 = p0[1], X2 = p1[0], X3 = p1[1];

        __builtin_amdgcn_sched_barrier(0);   // keep loads clustered above

        // ---- compute group ----
        float xv[16] = {X0.x, X0.y, X0.z, X0.w, X1.x, X1.y, X1.z, X1.w,
                        X2.x, X2.y, X2.z, X2.w, X3.x, X3.y, X3.z, X3.w};
        v16h a;
        #pragma unroll
        for (int u = 0; u < 16; ++u) a[u] = (_Float16)xv[u];
        #pragma unroll
        for (int t = 0; t < 8; ++t)
            c[t] = __builtin_amdgcn_wmma_f32_16x16x32_f16(false, a, false, bt[t],
                                                          (short)0, c[t], false, false);
    }
    #pragma unroll
    for (int t = 0; t < 8; ++t)
        #pragma unroll
        for (int r = 0; r < 8; ++r) {
            int row = rowBase + r + (hi ? 8 : 0);
            h[(size_t)row * F_OUT + t * 16 + m] = c[t][r];
        }
}

// ---------------------------------------------------------------------------
// Pack h [8192][128] f32 -> WMMA B-layout f16 tiles hB[jb][t][lane][v].
// ---------------------------------------------------------------------------
__global__ __launch_bounds__(256) void packB_kernel(const float* __restrict__ h,
                                                    unsigned int* __restrict__ hB) {
    int tid  = blockIdx.x * 256 + threadIdx.x;          // 524288 total
    int v    = tid & 7;
    int lane = (tid >> 3) & 31;
    int t    = (tid >> 8) & 7;
    int jb   = tid >> 11;
    int n    = lane & 15;
    int kb   = (lane >= 16) ? 8 : 0;
    int row  = 32 * jb + kb + 2 * v + (v >= 4 ? 8 : 0);
    int col  = 16 * t + n;
    _Float16 h0 = (_Float16)h[(size_t)row * F_OUT + col];
    _Float16 h1 = (_Float16)h[(size_t)(row + 1) * F_OUT + col];
    unsigned int u0 = (unsigned int)__builtin_bit_cast(unsigned short, h0);
    unsigned int u1 = (unsigned int)__builtin_bit_cast(unsigned short, h1);
    hB[tid] = u0 | (u1 << 16);
}

// ---------------------------------------------------------------------------
// f1 = h@a1, f2 = h@a2 (one row per thread), plus per-block max of f2.
// ---------------------------------------------------------------------------
__global__ __launch_bounds__(256) void f1f2_kernel(const float* __restrict__ h,
                                                   const float* __restrict__ a1,
                                                   const float* __restrict__ a2,
                                                   float* __restrict__ f1,
                                                   float* __restrict__ f2,
                                                   float* __restrict__ blockmax) {
    int row = blockIdx.x * 256 + threadIdx.x;
    const float4* hp  = (const float4*)(h + (size_t)row * F_OUT);
    const float4* w1p = (const float4*)a1;
    const float4* w2p = (const float4*)a2;
    float s1 = 0.f, s2 = 0.f;
    #pragma unroll
    for (int k = 0; k < F_OUT / 4; ++k) {
        float4 hv = hp[k], w1 = w1p[k], w2 = w2p[k];
        s1 += hv.x * w1.x + hv.y * w1.y + hv.z * w1.z + hv.w * w1.w;
        s2 += hv.x * w2.x + hv.y * w2.y + hv.z * w2.z + hv.w * w2.w;
    }
    f1[row] = s1;
    f2[row] = s2;

    float mx = s2;
    #pragma unroll
    for (int o = 16; o > 0; o >>= 1) mx = fmaxf(mx, __shfl_xor(mx, o, 32));
    __shared__ float sm[8];
    if ((threadIdx.x & 31) == 0) sm[threadIdx.x >> 5] = mx;
    __syncthreads();
    if (threadIdx.x == 0) {
        float m2 = sm[0];
        #pragma unroll
        for (int i = 1; i < 8; ++i) m2 = fmaxf(m2, sm[i]);
        blockmax[blockIdx.x] = m2;
    }
}

__global__ void gmax_kernel(const float* __restrict__ blockmax, float* __restrict__ gmax) {
    float v = blockmax[threadIdx.x];                    // 32 threads, 32 blockmax entries
    #pragma unroll
    for (int o = 16; o > 0; o >>= 1) v = fmaxf(v, __shfl_xor(v, o, 32));
    if (threadIdx.x == 0) *gmax = v;
}

// ---------------------------------------------------------------------------
// Fused masked-softmax-attention: single streaming pass over adj.
// Wave = 16 rows x 128 cols; probabilities built directly in f16 A-layout;
// 8x v_wmma_f32_16x16x32_f16 per 32-column step. j split JSPLIT ways.
// Manual 2x unroll with ping-pong register buffers (no rotation copies) +
// sched_barrier(0) between load group and compute group so B-tile loads
// cluster and waits are incremental.
// ---------------------------------------------------------------------------
__global__ __launch_bounds__(256) __attribute__((amdgpu_waves_per_eu(2)))
void attn_kernel(const int*   __restrict__ adj,
                 const float* __restrict__ f1,
                 const float* __restrict__ f2,
                 const float* __restrict__ gmaxp,
                 const v16h*  __restrict__ hB,
                 float* __restrict__ numP,
                 float* __restrict__ denP) {
    const int lane    = threadIdx.x & 31;
    const int wave    = threadIdx.x >> 5;
    const int rb      = blockIdx.x & 63;       // row block
    const int jh      = blockIdx.x >> 6;       // j chunk
    const int rowBase = rb * 128 + wave * 16;
    const int m       = lane & 15;
    const bool hi     = lane >= 16;
    const int kb0     = hi ? 8 : 0;

    const float gmax = *gmaxp;
    const float f1r  = f1[rowBase + m];
    const float xmax = f1r + gmax;
    const float Mi   = fmaxf(xmax, ALPHA * xmax);      // leaky_relu upper bound (monotone)

    v8f c[8] = {};
    float den = 0.f;

    const int  j0     = jh * JCHUNK;
    const int  j1     = j0 + JCHUNK;
    const int* adjRow = adj + (size_t)(rowBase + m) * N_NODES;

    // One pipeline phase: consume (CA*,CF*) for column block jc, and load
    // (XA*,XF*) for column block jn.  sched_barrier keeps the load group
    // (16 B-tile loads + 8 adj/f2 loads) clustered above the VALU+WMMA group.
    auto step = [&](int jc, int jn,
                    int4& CA0, int4& CA1, int4& CA2, int4& CA3,
                    float4& CF0, float4& CF1, float4& CF2, float4& CF3,
                    int4& XA0, int4& XA1, int4& XA2, int4& XA3,
                    float4& XF0, float4& XF1, float4& XF2, float4& XF3) {
        const int jb = jc >> 5;
        // ---- load group ----
        v16h bt[8];
        #pragma unroll
        for (int t = 0; t < 8; ++t) bt[t] = hB[(jb * 8 + t) * 32 + lane];

        const int4*   an0 = (const int4*)(adjRow + jn + kb0);
        const int4*   an1 = (const int4*)(adjRow + jn + kb0 + 16);
        XA0 = an0[0]; XA1 = an0[1]; XA2 = an1[0]; XA3 = an1[1];
        const float4* fn0 = (const float4*)(f2 + jn + kb0);
        const float4* fn1 = (const float4*)(f2 + jn + kb0 + 16);
        XF0 = fn0[0]; XF1 = fn0[1]; XF2 = fn1[0]; XF3 = fn1[1];

        __builtin_prefetch(adjRow + jc + 256, 0, 0);   // global_prefetch, ~8 blocks ahead
        __builtin_amdgcn_sched_barrier(0);

        // ---- compute group (current adj/f2 already resident) ----
        int   av[16] = {CA0.x, CA0.y, CA0.z, CA0.w, CA1.x, CA1.y, CA1.z, CA1.w,
                        CA2.x, CA2.y, CA2.z, CA2.w, CA3.x, CA3.y, CA3.z, CA3.w};
        float fv[16] = {CF0.x, CF0.y, CF0.z, CF0.w, CF1.x, CF1.y, CF1.z, CF1.w,
                        CF2.x, CF2.y, CF2.z, CF2.w, CF3.x, CF3.y, CF3.z, CF3.w};
        v16h  a;
        float psum = 0.f;
        #pragma unroll
        for (int u = 0; u < 16; ++u) {
            float s = f1r + fv[u];
            s = fmaxf(s, ALPHA * s);                   // leaky_relu
            float p = (av[u] > 0) ? __expf(s - Mi) : 0.0f;
            psum += p;
            a[u] = (_Float16)p;
        }
        den += psum;

        #pragma unroll
        for (int t = 0; t < 8; ++t)
            c[t] = __builtin_amdgcn_wmma_f32_16x16x32_f16(false, a, false, bt[t],
                                                          (short)0, c[t], false, false);
    };

    // ping-pong buffer sets (no rotation copies -> no end-of-iter waits)
    int4   A0, A1, A2, A3, B0, B1, B2, B3;
    float4 F0, F1, F2, F3, G0, G1, G2, G3;
    {   // prologue: load set A for j0
        const int4*   ap0 = (const int4*)(adjRow + j0 + kb0);
        const int4*   ap1 = (const int4*)(adjRow + j0 + kb0 + 16);
        A0 = ap0[0]; A1 = ap0[1]; A2 = ap1[0]; A3 = ap1[1];
        const float4* fp0 = (const float4*)(f2 + j0 + kb0);
        const float4* fp1 = (const float4*)(f2 + j0 + kb0 + 16);
        F0 = fp0[0]; F1 = fp0[1]; F2 = fp1[0]; F3 = fp1[1];
    }

    for (int j = j0; j < j1; j += 64) {
        // phase 0: consume A (block j), load B (block j+32)
        step(j, j + 32, A0, A1, A2, A3, F0, F1, F2, F3,
                        B0, B1, B2, B3, G0, G1, G2, G3);
        // phase 1: consume B (block j+32), load A (block j+64; wrap-clamped
        // on the final phase keeps addresses in-bounds, data unused)
        int jn2 = j + 64;
        if (jn2 >= j1) jn2 = j0;
        step(j + 32, jn2, B0, B1, B2, B3, G0, G1, G2, G3,
                          A0, A1, A2, A3, F0, F1, F2, F3);
    }

    float denFull = den + __shfl_xor(den, 16, 32);     // lanes m & m+16 jointly cover 32 cols
    if (!hi) denP[(size_t)jh * N_NODES + rowBase + m] = denFull;

    float* np = numP + (size_t)jh * ((size_t)N_NODES * F_OUT);
    #pragma unroll
    for (int t = 0; t < 8; ++t)
        #pragma unroll
        for (int r = 0; r < 8; ++r) {
            int row = rowBase + r + (hi ? 8 : 0);
            np[(size_t)row * F_OUT + t * 16 + m] = c[t][r];
        }
}

// ---------------------------------------------------------------------------
// Combine j-split partials: out = elu(sum(num)/sum(den)).
// ---------------------------------------------------------------------------
__global__ __launch_bounds__(256) void combine_kernel(const float* __restrict__ numP,
                                                      const float* __restrict__ denP,
                                                      float* __restrict__ out) {
    size_t tid = (size_t)blockIdx.x * 256 + threadIdx.x;   // N_NODES*F_OUT threads
    int row = (int)(tid >> 7);
    float num = 0.f, den = 0.f;
    #pragma unroll
    for (int jh = 0; jh < JSPLIT; ++jh) {
        den += denP[(size_t)jh * N_NODES + row];
        num += numP[(size_t)jh * ((size_t)N_NODES * F_OUT) + tid];
    }
    float v = num / den;
    out[tid] = (v > 0.f) ? v : (__expf(v) - 1.0f);         // ELU
}

// ---------------------------------------------------------------------------
extern "C" void kernel_launch(void* const* d_in, const int* in_sizes, int n_in,
                              void* d_out, int out_size, void* d_ws, size_t ws_size,
                              hipStream_t stream) {
    (void)in_sizes; (void)n_in; (void)out_size; (void)ws_size;
    const float* input = (const float*)d_in[0];   // [8192,256]
    const int*   adj   = (const int*)d_in[1];     // [8192,8192]
    const float* W     = (const float*)d_in[2];   // [256,128]
    const float* a1    = (const float*)d_in[3];   // [128]
    const float* a2    = (const float*)d_in[4];   // [128]
    float*       out   = (float*)d_out;           // [8192,128]

    char* ws = (char*)d_ws;
    size_t off = 0;
    auto carve = [&](size_t bytes) -> void* {
        void* p = (void*)(ws + off);
        off += (bytes + 255) & ~(size_t)255;
        return p;
    };
    float*        h     = (float*)carve((size_t)N_NODES * F_OUT * 4);      // 4 MB
    float*        f1    = (float*)carve(N_NODES * 4);
    float*        f2    = (float*)carve(N_NODES * 4);
    float*        bmax  = (float*)carve(32 * 4);
    float*        gmaxv = (float*)carve(256);
    unsigned int* WBp   = (unsigned int*)carve((size_t)F_IN * F_OUT * 2);  // 64 KB
    unsigned int* hBp   = (unsigned int*)carve((size_t)N_NODES * F_OUT * 2); // 2 MB
    float*        denP  = (float*)carve((size_t)JSPLIT * N_NODES * 4);
    float*        numP  = (float*)carve((size_t)JSPLIT * N_NODES * F_OUT * 4); // 16 MB

    packW_kernel<<<64, 256, 0, stream>>>(W, WBp);
    h_kernel<<<64, 256, 0, stream>>>(input, (const v16h*)WBp, h);
    packB_kernel<<<2048, 256, 0, stream>>>(h, hBp);
    f1f2_kernel<<<32, 256, 0, stream>>>(h, a1, a2, f1, f2, bmax);
    gmax_kernel<<<1, 32, 0, stream>>>(bmax, gmaxv);
    attn_kernel<<<64 * JSPLIT, 256, 0, stream>>>(adj, f1, f2, gmaxv,
                                                 (const v16h*)hBp, numP, denP);
    combine_kernel<<<(N_NODES * F_OUT) / 256, 256, 0, stream>>>(numP, denP, out);
}
